// Block_71528385347823
// MI455X (gfx1250) — matrix operand order repaired and checked by reading, not compiled, not run on hardware
//
#include <hip/hip_runtime.h>
#include <hip/hip_bf16.h>

// ---------------- types ----------------
typedef __bf16 bf16_t;
typedef __attribute__((ext_vector_type(16))) __bf16 v16bf;
typedef __attribute__((ext_vector_type(8)))  __bf16 v8bf;
typedef __attribute__((ext_vector_type(8)))  float  v8f;

#define TSEQ 2048
#define NEMBD 1024
#define NHEAD 16
#define HDIM 64
#define ROWS (2 * TSEQ)          // B*T = 4096

template <bool B> struct bool_c { static constexpr bool value = B; };

// ---------------- WMMA helpers ----------------
__device__ __forceinline__ v8f wmma_bf16(v16bf a, v16bf b, v8f c) {
    return __builtin_amdgcn_wmma_f32_16x16x32_bf16(
        false, a, false, b, (short)0, c, false, false);
}

// A fragment 16x32 (natural K order): lane = row (lane&15);
// e<8 -> K=k0+e+8*half, e>=8 -> K=k0+16+(e-8)+8*half
__device__ __forceinline__ v16bf ld_a_frag(const bf16_t* __restrict__ row, int k0, int half) {
    v8bf lo = *reinterpret_cast<const v8bf*>(row + k0 + 8 * half);
    v8bf hi = *reinterpret_cast<const v8bf*>(row + k0 + 16 + 8 * half);
    return __builtin_shufflevector(lo, hi, 0,1,2,3,4,5,6,7,8,9,10,11,12,13,14,15);
}

// B fragment 32x16 from [N,K] row-major: lane = col; K = k0 + e + 16*half (contiguous).
__device__ __forceinline__ v16bf ld_b_frag(const bf16_t* __restrict__ row, int k0, int half) {
    v8bf lo = *reinterpret_cast<const v8bf*>(row + k0 + 16 * half);
    v8bf hi = *reinterpret_cast<const v8bf*>(row + k0 + 16 * half + 8);
    return __builtin_shufflevector(lo, hi, 0,1,2,3,4,5,6,7,8,9,10,11,12,13,14,15);
}

__device__ __forceinline__ float gelu_tanh(float x) {
    const float c0 = 0.7978845608028654f, c1 = 0.044715f;
    float t = tanhf(c0 * (x + c1 * x * x * x));
    return 0.5f * x * (1.0f + t);
}

// ---------------- LayerNorm -> bf16 ----------------
__global__ void ln_bf16_kernel(const float* __restrict__ x,
                               const float* __restrict__ g,
                               const float* __restrict__ b,
                               bf16_t* __restrict__ out) {
    const int C = NEMBD;
    int row = blockIdx.x;
    const float* xr = x + (size_t)row * C;
    float v[4];
    float s = 0.f;
#pragma unroll
    for (int i = 0; i < 4; ++i) { v[i] = xr[threadIdx.x + i * 256]; s += v[i]; }
    __shared__ float red[256];
    red[threadIdx.x] = s; __syncthreads();
    for (int off = 128; off > 0; off >>= 1) {
        if (threadIdx.x < off) red[threadIdx.x] += red[threadIdx.x + off];
        __syncthreads();
    }
    float mu = red[0] * (1.0f / C);
    __syncthreads();
    float vs = 0.f;
#pragma unroll
    for (int i = 0; i < 4; ++i) { float d = v[i] - mu; vs += d * d; }
    red[threadIdx.x] = vs; __syncthreads();
    for (int off = 128; off > 0; off >>= 1) {
        if (threadIdx.x < off) red[threadIdx.x] += red[threadIdx.x + off];
        __syncthreads();
    }
    float rs = rsqrtf(red[0] * (1.0f / C) + 1e-5f);
#pragma unroll
    for (int i = 0; i < 4; ++i) {
        int c = threadIdx.x + i * 256;
        out[(size_t)row * C + c] = (bf16_t)((v[i] - mu) * rs * g[c] + b[c]);
    }
}

// ---------------- weight repack kernels ----------------
// WqkvT[n][c], n<1024:q, n<2048:k, n<3072:v ; col n -> head h=n/64, d=n%64
__global__ void pack_qkv_kernel(const float* __restrict__ Wq,
                                const float* __restrict__ Wk,
                                const float* __restrict__ Wv,
                                bf16_t* __restrict__ WqkvT) {
    int idx = blockIdx.x * 256 + threadIdx.x;   // over 3072*1024
    int c = idx & 1023;
    int n = idx >> 10;
    const float* W; int nn;
    if (n < 1024)       { W = Wq; nn = n; }
    else if (n < 2048)  { W = Wk; nn = n - 1024; }
    else                { W = Wv; nn = n - 2048; }
    int h = nn >> 6, d = nn & 63;
    WqkvT[idx] = (bf16_t)W[((size_t)h * NEMBD + c) * HDIM + d];
}

// out[n*K + k] = (bf16) in[k*N + n]
__global__ void transpose_bf16_kernel(const float* __restrict__ in,
                                      bf16_t* __restrict__ out, int N, int K) {
    size_t idx = (size_t)blockIdx.x * 256 + threadIdx.x;
    int k = (int)(idx % K);
    int n = (int)(idx / K);
    out[idx] = (bf16_t)in[(size_t)k * N + n];
}

// ---------------- generic WMMA GEMM ----------------
enum { EPI_QKV = 0, EPI_BIAS_RES = 1, EPI_BIAS_GELU = 2 };

// D[m,n] = sum_k A[m,k]*BT[n,k].  Block: 8 waves -> 32(M) x 256(N); wave -> 16 x 64.
// Register double-buffered K-loop: next step's fragments are issued before this
// step's WMMAs so loads overlap matrix ops.
template <int EPI>
__global__ void gemm_bf16_kernel(const bf16_t* __restrict__ A,
                                 const bf16_t* __restrict__ BT,
                                 const float* __restrict__ bias,
                                 const float* __restrict__ resid,
                                 float* __restrict__ outF,
                                 bf16_t* __restrict__ outB,
                                 bf16_t* __restrict__ outVT,
                                 int N, int K) {
    int tid  = threadIdx.x;
    int w    = tid >> 5;
    int lane = tid & 31;
    int half = lane >> 4;
    int ln   = lane & 15;
    int m0 = blockIdx.y * 32 + (w >> 2) * 16;
    int n0 = blockIdx.x * 256 + (w & 3) * 64;

    const bf16_t* arow = A + (size_t)(m0 + ln) * K;
    const bf16_t* brow[4];
#pragma unroll
    for (int j = 0; j < 4; ++j) brow[j] = BT + (size_t)(n0 + j * 16 + ln) * K;

    v8f acc[4] = {};

    // prologue: fragments for k0 = 0
    v16bf afc = ld_a_frag(arow, 0, half);
    v16bf bfc[4];
#pragma unroll
    for (int j = 0; j < 4; ++j) bfc[j] = ld_a_frag(brow[j], 0, half);

    for (int k0 = 32; k0 < K; k0 += 32) {
        // issue next step's loads first (overlap with current WMMAs)
        v16bf afn = ld_a_frag(arow, k0, half);
        v16bf bfn[4];
#pragma unroll
        for (int j = 0; j < 4; ++j) bfn[j] = ld_a_frag(brow[j], k0, half);
#pragma unroll
        for (int j = 0; j < 4; ++j) acc[j] = wmma_bf16(afc, bfc[j], acc[j]);
        afc = afn;
#pragma unroll
        for (int j = 0; j < 4; ++j) bfc[j] = bfn[j];
    }
    // epilogue step
#pragma unroll
    for (int j = 0; j < 4; ++j) acc[j] = wmma_bf16(afc, bfc[j], acc[j]);

#pragma unroll
    for (int j = 0; j < 4; ++j) {
#pragma unroll
        for (int r = 0; r < 8; ++r) {
            int m = m0 + r + 8 * half;
            int n = n0 + j * 16 + ln;
            float val = acc[j][r];
            if constexpr (EPI == EPI_QKV) {
                if (n < 2048) {
                    // fold the attention score scale C^-0.5 = 1/32 into q
                    if (n < 1024) val *= 0.03125f;
                    outB[(size_t)m * 2048 + n] = (bf16_t)val;   // q | k rows
                } else {                                        // v, transposed [B,H,D,T]
                    int nn = n - 2048;
                    int h = nn >> 6, d = nn & 63;
                    int b = m >> 11, t = m & (TSEQ - 1);
                    outVT[(((size_t)b * NHEAD + h) * HDIM + d) * TSEQ + t] = (bf16_t)val;
                }
            } else if constexpr (EPI == EPI_BIAS_RES) {
                size_t o = (size_t)m * N + n;
                outF[o] = val + bias[n] + resid[o];
            } else {
                size_t o = (size_t)m * N + n;
                outB[o] = (bf16_t)gelu_tanh(val + bias[n]);
            }
        }
    }
}

// ---------------- flash attention (transposed, per-wave 16-query tile) ----------------
// qk : [B*T, 2048] bf16 (cols 0..1023 = q*(1/32) at h*64+d, 1024..2047 = k)
// vT : [B,H,64,T]  bf16
// y  : [B*T, 1024] bf16 (concat heads)
//
// PV matmul uses a K-permutation pi (swap middle 8-groups of the 32-key block)
// applied to BOTH operands, which makes the P^T B-fragment exactly the values
// each lane already holds after softmax (no cross-lane ops), and makes the V^T
// A-fragment two 8-element loads at s = sb + 8*half and sb + 16 + 8*half (ld_a_frag form).
__global__ void attn_kernel(const bf16_t* __restrict__ qk,
                            const bf16_t* __restrict__ vT,
                            bf16_t* __restrict__ y) {
    int wave = blockIdx.x * 8 + (threadIdx.x >> 5);
    int lane = threadIdx.x & 31;
    int half = lane >> 4;
    int ln   = lane & 15;
    int qt = wave & 127;            // T/16 tiles
    int h  = (wave >> 7) & 15;
    int b  = wave >> 11;
    int qbase = qt * 16;

    // Q^T B-fragments (loaded once): lane = query column
    const bf16_t* qrow = qk + ((size_t)(b * TSEQ + qbase + ln)) * 2048 + h * HDIM;
    v16bf qf0 = ld_a_frag(qrow, 0, half);
    v16bf qf1 = ld_a_frag(qrow, 32, half);

    const bf16_t* kbase = qk + (size_t)b * TSEQ * 2048 + 1024 + h * HDIM;
    const bf16_t* vrow0 = vT + ((size_t)b * NHEAD + h) * HDIM * TSEQ + (size_t)ln * TSEQ;

    v8f o0 = {}, o1 = {}, o2 = {}, o3 = {};
    float Mx = -1e30f, L = 0.0f;
    int qg = qbase + ln;                 // this lane's query column (global)
    int nblk  = (qbase + 47) >> 5;       // total causal key blocks of 32
    int nfull = (qbase >= 32) ? (((qbase - 32) >> 5) + 1) : 0;  // blocks with no masking

    auto do_block = [&](int sb, auto masked_tag) {
        constexpr bool MASKED = decltype(masked_tag)::value;
        // ---- K-tile A-fragments (natural K order over d)
        const bf16_t* kr0 = kbase + (size_t)(sb + ln) * 2048;
        const bf16_t* kr1 = kr0 + (size_t)16 * 2048;
        v16bf ka00 = ld_a_frag(kr0, 0, half);
        v16bf ka10 = ld_a_frag(kr1, 0, half);
        v16bf ka01 = ld_a_frag(kr0, 32, half);
        v16bf ka11 = ld_a_frag(kr1, 32, half);
        // ---- V^T A-fragments in natural s order (matches pB): e<8 -> sb+8*half+e, e>=8 -> sb+16+8*half+(e-8)
        v16bf va0 = ld_a_frag(vrow0 + (size_t)(0 * 16) * TSEQ, sb, half);
        v16bf va1 = ld_a_frag(vrow0 + (size_t)(1 * 16) * TSEQ, sb, half);
        v16bf va2 = ld_a_frag(vrow0 + (size_t)(2 * 16) * TSEQ, sb, half);
        v16bf va3 = ld_a_frag(vrow0 + (size_t)(3 * 16) * TSEQ, sb, half);

        // ---- scores S^T = K x Q^T (two 16x16 tiles over s)
        v8f s0 = {}, s1 = {};
        s0 = wmma_bf16(ka00, qf0, s0);
        s1 = wmma_bf16(ka10, qf0, s1);
        s0 = wmma_bf16(ka01, qf1, s0);
        s1 = wmma_bf16(ka11, qf1, s1);

        // online softmax over s (per lane-pair column); scale already folded into q
        float p0[8], p1[8];
        float bm = -1e30f;
#pragma unroll
        for (int r = 0; r < 8; ++r) {
            float a = s0[r];
            float c = s1[r];
            if constexpr (MASKED) {
                int sA = sb + r + 8 * half;
                if (sA > qg)      a = -1e30f;
                if (sA + 16 > qg) c = -1e30f;
            }
            p0[r] = a; p1[r] = c;
            bm = fmaxf(bm, fmaxf(a, c));
        }
        bm = fmaxf(bm, __shfl_xor(bm, 16));
        float Mn   = fmaxf(Mx, bm);
        float corr = __expf(Mx - Mn);
        float ls = 0.f;
#pragma unroll
        for (int r = 0; r < 8; ++r) {
            p0[r] = __expf(p0[r] - Mn);
            p1[r] = __expf(p1[r] - Mn);
            ls += p0[r] + p1[r];
        }
        ls += __shfl_xor(ls, 16);
        L = L * corr + ls;
        Mx = Mn;
#pragma unroll
        for (int r = 0; r < 8; ++r) { o0[r] *= corr; o1[r] *= corr; o2[r] *= corr; o3[r] *= corr; }

        // P^T B-fragment in pi-permuted s order: exactly the values this lane holds.
        // half0: e<8 -> s=sb+e (p0), e>=8 -> s=sb+16+(e-8) (p1)
        // half1: e<8 -> s=sb+8+e (p0), e>=8 -> s=sb+24+(e-8) (p1)
        v16bf pB;
#pragma unroll
        for (int r = 0; r < 8; ++r) {
            pB[r]     = (__bf16)p0[r];
            pB[r + 8] = (__bf16)p1[r];
        }

        // O^T += V^T (A, permuted) x P^T (B, permuted) : 4 tiles covering d = 0..63
        o0 = wmma_bf16(va0, pB, o0);
        o1 = wmma_bf16(va1, pB, o1);
        o2 = wmma_bf16(va2, pB, o2);
        o3 = wmma_bf16(va3, pB, o3);
    };

    int blk = 0;
    for (; blk < nfull; ++blk) do_block(blk * 32, bool_c<false>{});   // fast path
    for (; blk < nblk;  ++blk) do_block(blk * 32, bool_c<true>{});    // masked tail

    float invL = 1.0f / L;
    bf16_t* yrow = y + (size_t)(b * TSEQ + qbase + ln) * NEMBD + h * HDIM;
#pragma unroll
    for (int r = 0; r < 8; ++r) {
        int d = r + 8 * half;
        yrow[d]      = (bf16_t)(o0[r] * invL);
        yrow[16 + d] = (bf16_t)(o1[r] * invL);
        yrow[32 + d] = (bf16_t)(o2[r] * invL);
        yrow[48 + d] = (bf16_t)(o3[r] * invL);
    }
}

// ---------------- host launch ----------------
extern "C" void kernel_launch(void* const* d_in, const int* in_sizes, int n_in,
                              void* d_out, int out_size, void* d_ws, size_t ws_size,
                              hipStream_t stream) {
    const float* x   = (const float*)d_in[0];
    const float* Wq  = (const float*)d_in[1];
    const float* Wk  = (const float*)d_in[2];
    const float* Wv  = (const float*)d_in[3];
    const float* Wo  = (const float*)d_in[4];
    const float* bo  = (const float*)d_in[5];
    const float* g1  = (const float*)d_in[6];
    const float* b1  = (const float*)d_in[7];
    const float* g2  = (const float*)d_in[8];
    const float* b2  = (const float*)d_in[9];
    const float* W1  = (const float*)d_in[10];
    const float* bb1 = (const float*)d_in[11];
    const float* W2  = (const float*)d_in[12];
    const float* bb2 = (const float*)d_in[13];
    float* out = (float*)d_out;

    char* ws = (char*)d_ws;
    size_t off = 0;
    auto alloc = [&](size_t bytes) -> void* {
        void* p = ws + off;
        off += (bytes + 255) & ~(size_t)255;
        return p;
    };
    bf16_t* h1    = (bf16_t*)alloc((size_t)ROWS * NEMBD * 2);
    bf16_t* WqkvT = (bf16_t*)alloc((size_t)3072 * 1024 * 2);
    bf16_t* WoT   = (bf16_t*)alloc((size_t)1024 * 1024 * 2);
    bf16_t* W1T   = (bf16_t*)alloc((size_t)4096 * 1024 * 2);
    bf16_t* W2T   = (bf16_t*)alloc((size_t)1024 * 4096 * 2);
    bf16_t* qkb   = (bf16_t*)alloc((size_t)ROWS * 2048 * 2);
    bf16_t* vTb   = (bf16_t*)alloc((size_t)2 * NHEAD * HDIM * TSEQ * 2);
    bf16_t* yb    = (bf16_t*)alloc((size_t)ROWS * NEMBD * 2);
    float*  x2    = (float*) alloc((size_t)ROWS * NEMBD * 4);
    bf16_t* h2    = (bf16_t*)alloc((size_t)ROWS * NEMBD * 2);
    bf16_t* a1    = (bf16_t*)alloc((size_t)ROWS * 4096 * 2);

    // 1) LN1
    ln_bf16_kernel<<<ROWS, 256, 0, stream>>>(x, g1, b1, h1);
    // 2) weight repack -> bf16 [N,K]
    pack_qkv_kernel<<<(3072 * 1024) / 256, 256, 0, stream>>>(Wq, Wk, Wv, WqkvT);
    transpose_bf16_kernel<<<(1024 * 1024) / 256, 256, 0, stream>>>(Wo, WoT, 1024, 1024);
    transpose_bf16_kernel<<<(1024 * 4096) / 256, 256, 0, stream>>>(W1, W1T, 4096, 1024);
    transpose_bf16_kernel<<<(4096 * 1024) / 256, 256, 0, stream>>>(W2, W2T, 1024, 4096);
    // 3) fused QKV projection (q scaled by 1/32; q,k row-major; v transposed)
    gemm_bf16_kernel<EPI_QKV><<<dim3(3072 / 256, ROWS / 32), 256, 0, stream>>>(
        h1, WqkvT, nullptr, nullptr, nullptr, qkb, vTb, 3072, 1024);
    // 4) causal flash attention
    attn_kernel<<<(2 * NHEAD * (TSEQ / 16)) / 8, 256, 0, stream>>>(qkb, vTb, yb);
    // 5) output projection + bias + residual -> x2 (f32)
    gemm_bf16_kernel<EPI_BIAS_RES><<<dim3(1024 / 256, ROWS / 32), 256, 0, stream>>>(
        yb, WoT, bo, x, x2, nullptr, nullptr, 1024, 1024);
    // 6) LN2
    ln_bf16_kernel<<<ROWS, 256, 0, stream>>>(x2, g2, b2, h2);
    // 7) MLP up + bias + GELU
    gemm_bf16_kernel<EPI_BIAS_GELU><<<dim3(4096 / 256, ROWS / 32), 256, 0, stream>>>(
        h2, W1T, bb1, nullptr, nullptr, a1, nullptr, 4096, 1024);
    // 8) MLP down + bias + residual -> d_out
    gemm_bf16_kernel<EPI_BIAS_RES><<<dim3(1024 / 256, ROWS / 32), 256, 0, stream>>>(
        a1, W2T, bb2, x2, out, nullptr, nullptr, 1024, 4096);
}
